// Generator_6528350290617
// MI455X (gfx1250) — matrix operand order repaired and checked by reading, not compiled, not run on hardware
//
#include <hip/hip_runtime.h>
#include <hip/hip_bf16.h>
#include <math.h>

typedef _Float16 f16;
typedef __attribute__((ext_vector_type(16))) _Float16 v16h;
typedef __attribute__((ext_vector_type(8)))  _Float16 v8h;
typedef __attribute__((ext_vector_type(8)))  float    v8f;

#define LSTRIDE 1032   // LDS spike row stride in f16 (16B aligned, bank-safe)
#define BETA 0.95f
#define THR  1.0f
#define NSTEP 25
#define BATCH 4096
#define ROWS  32       // batch rows per workgroup (2 WMMA M-tiles)

// Opaque scheduling fence: stops LLVM from hoisting (and then spilling)
// loop-invariant weight loads across timesteps/stages. Register-resident
// SSA state (membranes) is unaffected.
#define MEMFENCE() asm volatile("" ::: "memory")

// ---------------------------------------------------------------------------
// Precompute cur1 = x @ W1^T + b1   (constant across timesteps)  [4096 x 128]
// ---------------------------------------------------------------------------
__global__ void cur1_kernel(const float* __restrict__ x,
                            const float* __restrict__ W1,
                            const float* __restrict__ b1,
                            float* __restrict__ cur1) {
    int idx = blockIdx.x * blockDim.x + threadIdx.x;   // 4096*128 threads
    int b = idx >> 7;
    int n = idx & 127;
    float acc = b1[n];
    const float* xp = x  + b * 100;
    const float* wp = W1 + n * 100;
    #pragma unroll 4
    for (int k = 0; k < 100; ++k) acc = fmaf(xp[k], wp[k], acc);
    cur1[idx] = acc;
}

// ---------------------------------------------------------------------------
// Pack W (N x K, f32 row-major) into WMMA-B register image, f16.
// Tile (nT,kT) holds 512 f16: per lane L (0..31), 16 contiguous f16:
//   n = nT*16 + (L&15),  k = kT*32 + (L&16) + s,  s = 0..15
// ---------------------------------------------------------------------------
__global__ void pack_kernel(const float* __restrict__ W, f16* __restrict__ dst,
                            int K, int total) {
    int idx = blockIdx.x * blockDim.x + threadIdx.x;
    if (idx >= total) return;
    int s    = idx & 15;
    int lane = (idx >> 4) & 31;
    int tIdx = idx >> 9;
    int KT   = K >> 5;
    int kT   = tIdx % KT;
    int nT   = tIdx / KT;
    int k = kT * 32 + (lane & 16) + s;
    int n = nT * 16 + (lane & 15);
    dst[idx] = (f16)W[n * K + k];
}

// ---------------------------------------------------------------------------
// A-matrix fragment load from LDS spike buffer (16x32 f16, WMMA A layout).
// ---------------------------------------------------------------------------
__device__ __forceinline__ v16h load_a(const f16* __restrict__ in,
                                       int rowBase, int kT, int lane) {
    const int r    = lane & 15;
    const int aoff = (lane & 16) ? 16 : 0;
    const f16* ap = in + (rowBase + r) * LSTRIDE + kT * 32 + aoff;
    v8h lo = *(const v8h*)(ap);        // K = kb .. kb+7
    v8h hi = *(const v8h*)(ap + 8);    // K = kb+8 .. kb+15
    v16h a;
    #pragma unroll
    for (int e = 0; e < 8; ++e) { a[e] = lo[e]; a[8 + e] = hi[e]; }
    return a;
}

// ---------------------------------------------------------------------------
// Hidden LIF layer for a 32-row batch tile: each owned weight tile (B) is
// loaded ONCE into registers and reused for TWO row-tile WMMAs.
// m[] holds MAXT*2 membrane fragments: [i*2 + rowTile], WMMA C/D layout.
// ---------------------------------------------------------------------------
template <int NT, int KT, int MAXT>
__device__ __forceinline__ void lif_layer(const f16* __restrict__ Wp,
                                          const float* __restrict__ bias,
                                          const f16* __restrict__ in,
                                          f16* __restrict__ outSpk,
                                          v8f* m, int wave, int lane) {
    const int colHalf = lane & 15;
    const int rowOff  = (lane & 16) ? 8 : 0;
    #pragma unroll
    for (int i = 0; i < MAXT; ++i) {
        int nT = wave + i * 16;
        if ((NT % 16 != 0) && nT >= NT) break;           // wave-uniform
        v8f acc0 = {};
        v8f acc1 = {};
        const f16* wp = Wp + (size_t)(nT * KT) * 512 + lane * 16;
        #pragma unroll 2
        for (int kT = 0; kT < KT; ++kT) {
            v16h b  = *(const v16h*)(wp + (size_t)kT * 512);   // load once
            v16h a0 = load_a(in, 0,  kT, lane);
            v16h a1 = load_a(in, 16, kT, lane);
            acc0 = __builtin_amdgcn_wmma_f32_16x16x32_f16(
                false, a0, false, b, (short)0, acc0, false, false);
            acc1 = __builtin_amdgcn_wmma_f32_16x16x32_f16(
                false, a1, false, b, (short)0, acc1, false, false);
        }
        float bn = bias[nT * 16 + colHalf];
        #pragma unroll
        for (int rt = 0; rt < 2; ++rt) {
            v8f cur = (rt == 0) ? acc0 : acc1;
            v8f mv  = m[i * 2 + rt];
            #pragma unroll
            for (int v = 0; v < 8; ++v) {
                float reset = (mv[v] > THR) ? THR : 0.0f;    // reset from OLD mem
                float mn = BETA * mv[v] + (cur[v] + bn) - reset;
                mv[v] = mn;
                outSpk[(rt * 16 + rowOff + v) * LSTRIDE + nT * 16 + colHalf] =
                    (mn > THR) ? (f16)1.0f : (f16)0.0f;
            }
            m[i * 2 + rt] = mv;
        }
    }
}

// ---------------------------------------------------------------------------
// Output LIF layer (layer 5): GEMM + LIF, emits tanh(m5) to global.
// ---------------------------------------------------------------------------
template <int NT, int KT, int MAXT>
__device__ __forceinline__ void lif_layer_out(const f16* __restrict__ Wp,
                                              const float* __restrict__ bias,
                                              const f16* __restrict__ in,
                                              v8f* m,
                                              float* __restrict__ out,
                                              int row0, int wave, int lane) {
    const int colHalf = lane & 15;
    const int rowOff  = (lane & 16) ? 8 : 0;
    #pragma unroll
    for (int i = 0; i < MAXT; ++i) {
        int nT = wave + i * 16;
        if ((NT % 16 != 0) && nT >= NT) break;
        v8f acc0 = {};
        v8f acc1 = {};
        const f16* wp = Wp + (size_t)(nT * KT) * 512 + lane * 16;
        #pragma unroll 2
        for (int kT = 0; kT < KT; ++kT) {
            v16h b  = *(const v16h*)(wp + (size_t)kT * 512);
            v16h a0 = load_a(in, 0,  kT, lane);
            v16h a1 = load_a(in, 16, kT, lane);
            acc0 = __builtin_amdgcn_wmma_f32_16x16x32_f16(
                false, a0, false, b, (short)0, acc0, false, false);
            acc1 = __builtin_amdgcn_wmma_f32_16x16x32_f16(
                false, a1, false, b, (short)0, acc1, false, false);
        }
        float bn = bias[nT * 16 + colHalf];
        #pragma unroll
        for (int rt = 0; rt < 2; ++rt) {
            v8f cur = (rt == 0) ? acc0 : acc1;
            v8f mv  = m[i * 2 + rt];
            #pragma unroll
            for (int v = 0; v < 8; ++v) {
                float reset = (mv[v] > THR) ? THR : 0.0f;
                float mn = BETA * mv[v] + (cur[v] + bn) - reset;
                mv[v] = mn;
                out[(size_t)(row0 + rt * 16 + rowOff + v) * 784 + nT * 16 + colHalf] =
                    tanhf(mn);
            }
            m[i * 2 + rt] = mv;
        }
    }
}

// ---------------------------------------------------------------------------
// Main SNN kernel: one workgroup = 32 batch rows, 16 wave32s, all 25 steps.
// Membrane state lives in VGPRs; spikes ping-pong through LDS; each weight
// tile loaded once per step serves two row-tile WMMAs (halves L2 traffic).
// ---------------------------------------------------------------------------
__global__ __launch_bounds__(512) void snn_kernel(
    const float* __restrict__ cur1,
    const f16* __restrict__ W2p, const float* __restrict__ b2,
    const f16* __restrict__ W3p, const float* __restrict__ b3,
    const f16* __restrict__ W4p, const float* __restrict__ b4,
    const f16* __restrict__ W5p, const float* __restrict__ b5,
    float* __restrict__ out) {
    __shared__ __align__(16) f16 sA[ROWS * LSTRIDE];
    __shared__ __align__(16) f16 sB[ROWS * LSTRIDE];

    const int lane    = threadIdx.x & 31;
    const int wave    = threadIdx.x >> 5;
    const int row0    = blockIdx.x * ROWS;
    const int colHalf = lane & 15;
    const int rowOff  = (lane & 16) ? 8 : 0;

    // Layer 1: 8 column tiles x 2 row tiles = 16 tiles -> one per wave.
    const int l1rt = wave & 1;         // row tile
    const int l1n  = (wave >> 1) * 16 + colHalf;
    const float* c1p = cur1 + (size_t)(row0 + l1rt * 16 + rowOff) * 128 + l1n;

    v8f m1    = {};
    v8f m2[2] = {};
    v8f m3[4] = {};
    v8f m4[8] = {};
    v8f m5[8] = {};

    for (int t = 0; t < NSTEP; ++t) {
        MEMFENCE();
        // ---- layer 1: LIF on constant current (re-read from L2), spikes -> sA
        {
            #pragma unroll
            for (int v = 0; v < 8; ++v) {
                float c1 = c1p[(size_t)v * 128];
                float reset = (m1[v] > THR) ? THR : 0.0f;
                float mn = BETA * m1[v] + c1 - reset;
                m1[v] = mn;
                sA[(l1rt * 16 + rowOff + v) * LSTRIDE + l1n] =
                    (mn > THR) ? (f16)1.0f : (f16)0.0f;
            }
        }
        __syncthreads();
        MEMFENCE();
        // ---- layer 2: 128 -> 256   (sA -> sB)
        lif_layer<16, 4, 1>(W2p, b2, sA, sB, m2, wave, lane);
        __syncthreads();
        MEMFENCE();
        // ---- layer 3: 256 -> 512   (sB -> sA)
        lif_layer<32, 8, 2>(W3p, b3, sB, sA, m3, wave, lane);
        __syncthreads();
        MEMFENCE();
        // ---- layer 4: 512 -> 1024  (sA -> sB)
        lif_layer<64, 16, 4>(W4p, b4, sA, sB, m4, wave, lane);
        __syncthreads();
        MEMFENCE();
        // ---- layer 5: 1024 -> 784, tanh(m5) -> out[t]
        lif_layer_out<49, 32, 4>(W5p, b5, sB, m5,
                                 out + (size_t)t * BATCH * 784, row0, wave, lane);
        __syncthreads();
    }
}

// ---------------------------------------------------------------------------
extern "C" void kernel_launch(void* const* d_in, const int* in_sizes, int n_in,
                              void* d_out, int out_size, void* d_ws, size_t ws_size,
                              hipStream_t stream) {
    const float* x  = (const float*)d_in[0];
    const float* W1 = (const float*)d_in[1];
    const float* b1 = (const float*)d_in[2];
    const float* W2 = (const float*)d_in[3];
    const float* b2 = (const float*)d_in[4];
    const float* W3 = (const float*)d_in[5];
    const float* b3 = (const float*)d_in[6];
    const float* W4 = (const float*)d_in[7];
    const float* b4 = (const float*)d_in[8];
    const float* W5 = (const float*)d_in[9];
    const float* b5 = (const float*)d_in[10];

    char* ws = (char*)d_ws;
    float* cur1 = (float*)ws;                       // 4096*128 f32 = 2 MB
    f16* W2p = (f16*)(ws + (size_t)BATCH * 128 * 4);
    f16* W3p = W2p + 256 * 128;
    f16* W4p = W3p + 512 * 256;
    f16* W5p = W4p + 1024 * 512;

    cur1_kernel<<<(BATCH * 128) / 256, 256, 0, stream>>>(x, W1, b1, cur1);
    pack_kernel<<<(256 * 128) / 256, 256, 0, stream>>>(W2, W2p, 128, 256 * 128);
    pack_kernel<<<(512 * 256) / 256, 256, 0, stream>>>(W3, W3p, 256, 512 * 256);
    pack_kernel<<<(1024 * 512) / 256, 256, 0, stream>>>(W4, W4p, 512, 1024 * 512);
    pack_kernel<<<(784 * 1024) / 256, 256, 0, stream>>>(W5, W5p, 1024, 784 * 1024);

    snn_kernel<<<BATCH / ROWS, 512, 0, stream>>>(cur1, W2p, b2, W3p, b3,
                                                 W4p, b4, W5p, b5, (float*)d_out);
}
